// MultiHeadAttention_309237645580
// MI455X (gfx1250) — matrix operand order repaired and checked by reading, not compiled, not run on hardware
//
#include <hip/hip_runtime.h>

typedef __attribute__((ext_vector_type(16))) __bf16 v16bf;
typedef __attribute__((ext_vector_type(8)))  __bf16 v8bf;
typedef __attribute__((ext_vector_type(8)))  float  v8f;
typedef __attribute__((ext_vector_type(4)))  unsigned int u32x4;
typedef __attribute__((ext_vector_type(8)))  unsigned int u32x8;

#define B_  4
#define S_  2048
#define H_  16
#define DK_ 64
#define DM_ 1024   // N_HEAD * D_K
// total tokens = B_*S_ = 8192

#define SC_BYTES   (16 * S_ * 4)       // 16 x 2048 fp32 score strip = 128 KB
#define QTILE_OFF  SC_BYTES            // Q tile (16 x 64 bf16) behind it
#define SMEM_BYTES (SC_BYTES + 16 * DK_ * 2)

// ---------------------------------------------------------------------------
// WMMA helpers (CDNA5 bf16 16x16x32, fp32 accumulate)
// ---------------------------------------------------------------------------
__device__ __forceinline__ v8f wmma_bf16(v16bf a, v16bf b, v8f c) {
  // (neg_a, A, neg_b, B, c_mod, C, reuse_a, reuse_b)
  return __builtin_amdgcn_wmma_f32_16x16x32_bf16(false, a, false, b, (short)0, c,
                                                 false, false);
}

__device__ __forceinline__ v8f zero8() {
  v8f z;
#pragma unroll
  for (int i = 0; i < 8; ++i) z[i] = 0.0f;
  return z;
}

// A fragment: 16x32 bf16 tile at (m0,k0) of row-major A, leading dim lda.
// ISA layout: lanes 0-15 -> M=lane, elems 0..7 = K k0..k0+7, 8..15 = K k0+16..+23
//             lanes 16-31 -> M=lane-16, K-halves shifted by 8.
__device__ __forceinline__ v16bf load_a16(const __bf16* __restrict__ A, int lda,
                                          int m0, int k0, int lane) {
  const int mr = lane & 15;
  const int kh = (lane & 16) ? 8 : 0;
  const __bf16* p = A + (size_t)(m0 + mr) * lda + k0 + kh;
  v8bf lo = *(const v8bf*)(p);
  v8bf hi = *(const v8bf*)(p + 16);
  v16bf r;
#pragma unroll
  for (int i = 0; i < 8; ++i) { r[i] = lo[i]; r[i + 8] = hi[i]; }
  return r;
}

// B fragment: 32x16 (K x N) tile. Bt is stored N-major (Bt[n][k], leading dim
// ldb): lanes 0-15 elems 0..7 = K k0..k0+7, 8..15 = K k0+16..+23;
// lanes 16-31 K-halves shifted by 8, N = lane&15.
__device__ __forceinline__ v16bf load_b16(const __bf16* __restrict__ Bt, int ldb,
                                          int n0, int k0, int lane) {
  const int n = lane & 15;
  const int kh = (lane & 16) ? 8 : 0;
  const __bf16* p = Bt + (size_t)(n0 + n) * ldb + k0 + kh;
  v8bf lo = *(const v8bf*)(p);
  v8bf hi = *(const v8bf*)(p + 16);
  v16bf r;
#pragma unroll
  for (int i = 0; i < 8; ++i) { r[i] = lo[i]; r[i + 8] = hi[i]; }
  return r;
}

// ---------------------------------------------------------------------------
// Elementwise conversion kernels
// ---------------------------------------------------------------------------
__global__ void cvt_f32_bf16_kernel(const float* __restrict__ x,
                                    __bf16* __restrict__ y, int n) {
  for (int i = blockIdx.x * blockDim.x + threadIdx.x; i < n;
       i += gridDim.x * blockDim.x)
    y[i] = (__bf16)x[i];
}

// Wt[n][k] = W[k][n], fp32 -> bf16 (1024x1024)
__global__ void wtrans_kernel(const float* __restrict__ W, __bf16* __restrict__ Wt) {
  const int k = blockIdx.x * 16 + threadIdx.x;
  const int n = blockIdx.y * 16 + threadIdx.y;
  Wt[(size_t)n * DM_ + k] = (__bf16)W[(size_t)k * DM_ + n];
}

// ---------------------------------------------------------------------------
// Projection GEMM: Y[M=8192, N=1024] = A[M,K=1024] @ W (W given as Wt[n][k]).
// Block = 128 threads (4 waves stacked in M), tile 64(M) x 64(N).
// Register double-buffered k-loop so global loads overlap the WMMA pipe.
// vtranspose!=0 stores V head-transposed [B,H,64,S] for the P@V GEMM.
// ---------------------------------------------------------------------------
__global__ void proj_gemm_kernel(const __bf16* __restrict__ A,
                                 const __bf16* __restrict__ Bt,
                                 __bf16* __restrict__ Y, float scale,
                                 int vtranspose) {
  const int lane = threadIdx.x & 31;
  const int wave = threadIdx.x >> 5;
  const int m0 = blockIdx.x * 64 + wave * 16;
  const int n0 = blockIdx.y * 64;
  v8f acc[4];
#pragma unroll
  for (int j = 0; j < 4; ++j) acc[j] = zero8();

  v16bf a_cur = load_a16(A, DM_, m0, 0, lane);
  v16bf b_cur[4];
#pragma unroll
  for (int j = 0; j < 4; ++j) b_cur[j] = load_b16(Bt, DM_, n0 + j * 16, 0, lane);

  for (int k0 = 0; k0 < DM_; k0 += 32) {
    const int kn = (k0 + 32 < DM_) ? k0 + 32 : k0;  // last iter reloads (dead)
    v16bf a_nxt = load_a16(A, DM_, m0, kn, lane);
    v16bf b_nxt[4];
#pragma unroll
    for (int j = 0; j < 4; ++j) b_nxt[j] = load_b16(Bt, DM_, n0 + j * 16, kn, lane);
    if (k0 + 64 < DM_)  // CDNA5 global_prefetch_b8 of the A line after next
      __builtin_prefetch(A + (size_t)(m0 + (lane & 15)) * DM_ + k0 + 64, 0, 0);
#pragma unroll
    for (int j = 0; j < 4; ++j) acc[j] = wmma_bf16(a_cur, b_cur[j], acc[j]);
    a_cur = a_nxt;
#pragma unroll
    for (int j = 0; j < 4; ++j) b_cur[j] = b_nxt[j];
  }

  const int nn = lane & 15;
  const int mb = (lane & 16) ? 8 : 0;
#pragma unroll
  for (int j = 0; j < 4; ++j) {
#pragma unroll
    for (int r = 0; r < 8; ++r) {
      const float v = acc[j][r] * scale;
      const int m = m0 + mb + r;
      const int n = n0 + j * 16 + nn;
      if (!vtranspose) {
        Y[(size_t)m * DM_ + n] = (__bf16)v;
      } else {
        const int bb = m >> 11;         // m / S_
        const int s  = m & (S_ - 1);
        const int h  = n >> 6;          // n / DK_
        const int d  = n & (DK_ - 1);
        Y[((size_t)((bb * H_ + h) * DK_ + d)) * S_ + s] = (__bf16)v;
      }
    }
  }
}

// ---------------------------------------------------------------------------
// Output projection + residual: preLN = ctx @ w_fc + q   (fp32 out)
// ---------------------------------------------------------------------------
__global__ void outproj_gemm_kernel(const __bf16* __restrict__ A,
                                    const __bf16* __restrict__ Bt,
                                    const float* __restrict__ residual,
                                    float* __restrict__ Y) {
  const int lane = threadIdx.x & 31;
  const int wave = threadIdx.x >> 5;
  const int m0 = blockIdx.x * 64 + wave * 16;
  const int n0 = blockIdx.y * 64;
  v8f acc[4];
#pragma unroll
  for (int j = 0; j < 4; ++j) acc[j] = zero8();

  v16bf a_cur = load_a16(A, DM_, m0, 0, lane);
  v16bf b_cur[4];
#pragma unroll
  for (int j = 0; j < 4; ++j) b_cur[j] = load_b16(Bt, DM_, n0 + j * 16, 0, lane);

  for (int k0 = 0; k0 < DM_; k0 += 32) {
    const int kn = (k0 + 32 < DM_) ? k0 + 32 : k0;
    v16bf a_nxt = load_a16(A, DM_, m0, kn, lane);
    v16bf b_nxt[4];
#pragma unroll
    for (int j = 0; j < 4; ++j) b_nxt[j] = load_b16(Bt, DM_, n0 + j * 16, kn, lane);
#pragma unroll
    for (int j = 0; j < 4; ++j) acc[j] = wmma_bf16(a_cur, b_cur[j], acc[j]);
    a_cur = a_nxt;
#pragma unroll
    for (int j = 0; j < 4; ++j) b_cur[j] = b_nxt[j];
  }

  const int nn = lane & 15;
  const int mb = (lane & 16) ? 8 : 0;
#pragma unroll
  for (int j = 0; j < 4; ++j)
#pragma unroll
    for (int r = 0; r < 8; ++r) {
      const size_t idx = (size_t)(m0 + mb + r) * DM_ + n0 + j * 16 + nn;
      Y[idx] = acc[j][r] + residual[idx];
    }
}

// ---------------------------------------------------------------------------
// Fused attention: one workgroup (256 thr = 8 waves) owns a (b, h, 16-row q
// strip). The 16x64 bf16 Q tile is DMA'd into LDS once by the Tensor Data
// Mover (tensor_load_to_lds, waited with s_wait_tensorcnt); the full
// 16 x 2048 fp32 score strip lives in LDS (128 KB of the 320 KB WGP LDS).
// Phase 1: WMMA Q.K^T + mask -> LDS. Phase 2: in-LDS softmax, single
// non-temporal write of the normalized attn matrix. Phase 3: WMMA P@V with a
// double-buffered V stream, probs read straight from LDS.
// ---------------------------------------------------------------------------
__global__ void attn_kernel(const __bf16* __restrict__ qh,
                            const __bf16* __restrict__ kh,
                            const __bf16* __restrict__ vt,
                            const int* __restrict__ mask,
                            float* __restrict__ attn_out,
                            __bf16* __restrict__ ctx) {
  extern __shared__ char smem[];
  float* sc = (float*)smem;                              // [16][S_]
  const __bf16* qlds = (const __bf16*)(smem + QTILE_OFF);  // [16][64]

  const int lane = threadIdx.x & 31;
  const int wave = threadIdx.x >> 5;
  const int q0 = blockIdx.x * 16;
  const int h = blockIdx.y;
  const int b = blockIdx.z;
  const __bf16* qrow = qh + (size_t)b * S_ * DM_ + h * DK_;
  const __bf16* krow = kh + (size_t)b * S_ * DM_ + h * DK_;
  const int nn = lane & 15;
  const int mb = (lane & 16) ? 8 : 0;

  // ---- TDM: DMA the 16x64 bf16 Q tile into LDS (one wave issues it) ----
  if (threadIdx.x < 32) {
    const unsigned long long ga =
        (unsigned long long)(size_t)(qrow + (size_t)q0 * DM_);
    u32x4 g0;
    g0[0] = 1u;                                        // count=1, user mode
    g0[1] = (unsigned)QTILE_OFF;                       // lds_addr (bytes)
    g0[2] = (unsigned)(ga & 0xffffffffu);              // global_addr[31:0]
    g0[3] = (unsigned)((ga >> 32) & 0x01ffffffu)       // global_addr[56:32]
            | (2u << 30);                              // type = 2 ("image")
    u32x8 g1;
    g1[0] = 1u << 16;    // workgroup_mask=0, data_size=1 (2 bytes)
    g1[1] = 64u << 16;   // tensor_dim0 = 64      (bits 79:48)
    g1[2] = 16u << 16;   // tensor_dim1 = 16      (bits 111:80)
    g1[3] = 64u << 16;   // tile_dim0   = 64      (bits 127:112)
    g1[4] = 16u;         // tile_dim1   = 16      (bits 143:128), tile_dim2=0
    g1[5] = (unsigned)DM_;  // tensor_dim0_stride = 1024 (bits 207:160)
    g1[6] = 0u;
    g1[7] = 0u;
    asm volatile("tensor_load_to_lds %0, %1" :: "s"(g0), "s"(g1) : "memory");
    __builtin_amdgcn_s_wait_tensorcnt(0);
  }
  __syncthreads();

  // Hoisted Q A-fragments (k0 = 0 and 32), read once from LDS per wave.
  const v16bf qa0 = load_a16(qlds, DK_, 0, 0, lane);
  const v16bf qa1 = load_a16(qlds, DK_, 0, 32, lane);

  // ---- Phase 1: scores = (Q/sqrt(d)) K^T  (scale folded into qh) ----
  for (int kt = wave; kt < S_ / 16; kt += 8) {
    const int key0 = kt * 16;
    v8f acc = zero8();
    v16bf b0 = load_b16(krow, DM_, key0, 0, lane);
    v16bf b1 = load_b16(krow, DM_, key0, 32, lane);
    acc = wmma_bf16(qa0, b0, acc);
    acc = wmma_bf16(qa1, b1, acc);
#pragma unroll
    for (int r = 0; r < 8; ++r) {
      const int qq = q0 + mb + r;
      const int key = key0 + nn;
      const int mv =
          __builtin_nontemporal_load(mask + ((size_t)b * S_ + qq) * S_ + key);
      sc[(mb + r) * S_ + key] = mv ? acc[r] : -1.0e9f;
    }
  }
  __syncthreads();

  // ---- Phase 2: softmax per row (wave w owns rows 2w, 2w+1) ----
#pragma unroll
  for (int rr = 0; rr < 2; ++rr) {
    const int r = wave * 2 + rr;
    float* row = sc + r * S_;
    float mx = -3.4e38f;
    for (int i = lane; i < S_; i += 32) mx = fmaxf(mx, row[i]);
#pragma unroll
    for (int o = 16; o; o >>= 1) mx = fmaxf(mx, __shfl_xor(mx, o));
    float sum = 0.0f;
    for (int i = lane; i < S_; i += 32) {
      const float e = __expf(row[i] - mx);
      row[i] = e;
      sum += e;
    }
#pragma unroll
    for (int o = 16; o; o >>= 1) sum += __shfl_xor(sum, o);
    const float inv = 1.0f / sum;
    float* aout = attn_out + ((size_t)(b * H_ + h) * S_ + q0 + r) * S_;
    for (int i = lane; i < S_; i += 32) {
      const float pv = row[i] * inv;
      row[i] = pv;
      __builtin_nontemporal_store(pv, aout + i);  // 1.07 GB stream: NT hint
    }
  }
  __syncthreads();

  // ---- Phase 3: ctx[16,64] = P[16,2048] @ V[2048,64] (waves 0..3) ----
  if (wave < 4) {
    const int n0 = wave * 16;
    const __bf16* vbase = vt + (size_t)(b * H_ + h) * DK_ * S_;  // [64][S_]
    v8f acc = zero8();
    const int mr = lane & 15;
    const int kh0 = (lane & 16) ? 8 : 0;

    auto load_pa = [&](int k0) -> v16bf {
      const float* p = sc + mr * S_ + k0 + kh0;
      v8f lo = *(const v8f*)(p);
      v8f hi = *(const v8f*)(p + 16);
      v16bf a;
#pragma unroll
      for (int i = 0; i < 8; ++i) {
        a[i] = (__bf16)lo[i];
        a[i + 8] = (__bf16)hi[i];
      }
      return a;
    };

    v16bf b_cur = load_b16(vbase, S_, n0, 0, lane);
    for (int k0 = 0; k0 < S_; k0 += 32) {
      const int kn = (k0 + 32 < S_) ? k0 + 32 : k0;
      v16bf b_nxt = load_b16(vbase, S_, n0, kn, lane);  // prefetch next V frag
      v16bf a = load_pa(k0);
      acc = wmma_bf16(a, b_cur, acc);
      b_cur = b_nxt;
    }
#pragma unroll
    for (int r = 0; r < 8; ++r)
      ctx[((size_t)b * S_ + q0 + mb + r) * DM_ + h * DK_ + n0 + nn] =
          (__bf16)acc[r];
  }
}

// ---------------------------------------------------------------------------
// LayerNorm over last dim (1024), one block (8 waves) per token row.
// ---------------------------------------------------------------------------
__global__ void ln_kernel(const float* __restrict__ x,
                          const float* __restrict__ gamma,
                          const float* __restrict__ beta,
                          float* __restrict__ out) {
  const int row = blockIdx.x;
  const float* p = x + (size_t)row * DM_;
  float s = 0.0f, s2 = 0.0f;
  for (int i = threadIdx.x; i < DM_; i += blockDim.x) {
    const float t = p[i];
    s += t;
    s2 += t * t;
  }
#pragma unroll
  for (int o = 16; o; o >>= 1) {
    s += __shfl_xor(s, o);
    s2 += __shfl_xor(s2, o);
  }
  __shared__ float as[8], as2[8];
  const int w = threadIdx.x >> 5;
  if ((threadIdx.x & 31) == 0) { as[w] = s; as2[w] = s2; }
  __syncthreads();
  s = 0.0f; s2 = 0.0f;
#pragma unroll
  for (int j = 0; j < 8; ++j) { s += as[j]; s2 += as2[j]; }
  const float mean = s * (1.0f / DM_);
  const float var = s2 * (1.0f / DM_) - mean * mean;
  const float inv = rsqrtf(var + 1e-6f);
  float* o = out + (size_t)row * DM_;
  for (int i = threadIdx.x; i < DM_; i += blockDim.x)
    o[i] = (p[i] - mean) * inv * gamma[i] + beta[i];
}

// ---------------------------------------------------------------------------
// Host launcher
// ---------------------------------------------------------------------------
extern "C" void kernel_launch(void* const* d_in, const int* in_sizes, int n_in,
                              void* d_out, int out_size, void* d_ws, size_t ws_size,
                              hipStream_t stream) {
  (void)in_sizes; (void)n_in; (void)out_size; (void)ws_size;

  const float* q     = (const float*)d_in[0];
  const float* k     = (const float*)d_in[1];
  const float* v     = (const float*)d_in[2];
  const int*   mask  = (const int*)d_in[3];
  const float* w_qs  = (const float*)d_in[4];
  const float* w_ks  = (const float*)d_in[5];
  const float* w_vs  = (const float*)d_in[6];
  const float* w_fc  = (const float*)d_in[7];
  const float* gamma = (const float*)d_in[8];
  const float* beta  = (const float*)d_in[9];

  char* ws = (char*)d_ws;
  size_t off = 0;
  auto alloc = [&](size_t bytes) -> void* {
    void* p = ws + off;
    off = (off + bytes + 255) & ~(size_t)255;
    return p;
  };

  const size_t NTOK = (size_t)B_ * S_;       // 8192 tokens
  const size_t NELT = NTOK * DM_;            // 8,388,608 activations

  __bf16* q_bf  = (__bf16*)alloc(NELT * 2);
  __bf16* k_bf  = (__bf16*)alloc(NELT * 2);
  __bf16* v_bf  = (__bf16*)alloc(NELT * 2);
  __bf16* wq_t  = (__bf16*)alloc((size_t)DM_ * DM_ * 2);
  __bf16* wk_t  = (__bf16*)alloc((size_t)DM_ * DM_ * 2);
  __bf16* wv_t  = (__bf16*)alloc((size_t)DM_ * DM_ * 2);
  __bf16* wfc_t = (__bf16*)alloc((size_t)DM_ * DM_ * 2);
  __bf16* qhb   = (__bf16*)alloc(NELT * 2);  // [B,S,H*Dk], pre-scaled 1/sqrt(Dk)
  __bf16* khb   = (__bf16*)alloc(NELT * 2);  // [B,S,H*Dk]
  __bf16* vh_t  = (__bf16*)alloc(NELT * 2);  // [B,H,Dk,S]
  __bf16* ctx   = (__bf16*)alloc(NELT * 2);  // [B,S,H*Dv]
  float*  preln = (float*)alloc(NELT * 4);   // pre-LN fp32

  float* out_ptr  = (float*)d_out;                 // [B,S,D_MODEL]
  float* attn_ptr = (float*)d_out + NELT;          // [B,H,S,S]

  // 1) convert activations to bf16
  cvt_f32_bf16_kernel<<<4096, 256, 0, stream>>>(q, q_bf, (int)NELT);
  cvt_f32_bf16_kernel<<<4096, 256, 0, stream>>>(k, k_bf, (int)NELT);
  cvt_f32_bf16_kernel<<<4096, 256, 0, stream>>>(v, v_bf, (int)NELT);

  // 2) transpose+convert weights (N-major for contiguous B-fragments)
  dim3 tg(64, 64), tb(16, 16);
  wtrans_kernel<<<tg, tb, 0, stream>>>(w_qs, wq_t);
  wtrans_kernel<<<tg, tb, 0, stream>>>(w_ks, wk_t);
  wtrans_kernel<<<tg, tb, 0, stream>>>(w_vs, wv_t);
  wtrans_kernel<<<tg, tb, 0, stream>>>(w_fc, wfc_t);

  // 3) QKV projection GEMMs (WMMA). Q gets 1/sqrt(Dk) folded in; V stored
  //    head-transposed for the P@V stage.
  dim3 gg(NTOK / 64, DM_ / 64);  // (128, 16)
  proj_gemm_kernel<<<gg, 128, 0, stream>>>(q_bf, wq_t, qhb, 0.125f, 0);
  proj_gemm_kernel<<<gg, 128, 0, stream>>>(k_bf, wk_t, khb, 1.0f, 0);
  proj_gemm_kernel<<<gg, 128, 0, stream>>>(v_bf, wv_t, vh_t, 1.0f, 1);

  // 4) fused attention (TDM Q-tile -> scores -> softmax -> attn write -> P@V)
  attn_kernel<<<dim3(S_ / 16, H_, B_), 256, SMEM_BYTES, stream>>>(
      qhb, khb, vh_t, mask, attn_ptr, ctx);

  // 5) output projection + residual
  outproj_gemm_kernel<<<gg, 128, 0, stream>>>(ctx, wfc_t, q, preln);

  // 6) LayerNorm -> final output
  ln_kernel<<<NTOK, 256, 0, stream>>>(preln, gamma, beta, out_ptr);
}
